// BiMambaMSM_87196426043875
// MI455X (gfx1250) — matrix operand order repaired and checked by reading, not compiled, not run on hardware
//
#include <hip/hip_runtime.h>

typedef _Float16 f16;
typedef __attribute__((ext_vector_type(16))) _Float16 v16h;
typedef __attribute__((ext_vector_type(8)))  _Float16 v8h;
typedef __attribute__((ext_vector_type(8)))  float    v8f;

constexpr int kDModel  = 512;
constexpr int kDState  = 16;
constexpr int kDConv   = 7;
constexpr int kHeadDim = 64;
constexpr int kDInner  = 1024;
constexpr int kNHeads  = 16;
constexpr int kConvDim = 1056;   // D_INNER + 2*D_STATE
constexpr int kDProj   = 2096;   // 2*D_INNER + 2*D_STATE + NHEADS
constexpr int kBatch   = 2;
constexpr int kSeq     = 1024;
constexpr int kFeat    = 80;
constexpr int kTokens  = kBatch * kSeq;   // 2048
constexpr float kEps   = 1e-5f;

__device__ __forceinline__ float siluf(float x) { return x / (1.0f + __expf(-x)); }

// ---------------------------------------------------------------------------
// Operand packing: f32 -> f16 with K padded to Kp (zero fill).
// ---------------------------------------------------------------------------
__global__ __launch_bounds__(256) void pack_rowmajor_f16(
    const float* __restrict__ A, f16* __restrict__ Ah, int K, int Kp, int total)
{
    int idx = blockIdx.x * 256 + threadIdx.x;
    if (idx >= total) return;
    int k = idx % Kp, m = idx / Kp;
    Ah[idx] = (k < K) ? (f16)A[(size_t)m * K + k] : (f16)0.f;
}

__global__ __launch_bounds__(256) void pack_transpose_f16(
    const float* __restrict__ B, f16* __restrict__ Bt, int K, int N, int Kp, int total)
{
    int idx = blockIdx.x * 256 + threadIdx.x;
    if (idx >= total) return;
    int k = idx % Kp, n = idx / Kp;
    Bt[idx] = (k < K) ? (f16)B[(size_t)k * N + n] : (f16)0.f;
}

// ---------------------------------------------------------------------------
// WMMA GEMM: C[M,N] (+)= A[M,K] * B[K,N] (+bias[N])
//   Ah: f16 [M][Kp] row-major (Kp % 32 == 0, zero padded)
//   Bt: f16 [N][Kp] (B transposed at pack time)
//   Each wave owns a 64(M) x 16(N) strip: 4 accumulators share one B fragment
//   -> 4 v_wmma per 32-wide K step, all fragment loads contiguous 128-bit.
//   revL != 0: A row (b,t) read from (b, revL-1-t)  (backward direction).
// ---------------------------------------------------------------------------
__global__ __launch_bounds__(256) void wmma_gemm(
    const f16* __restrict__ Ah, const f16* __restrict__ Bt,
    const float* __restrict__ bias, float* __restrict__ C,
    int N, int Kp, int revL, int accumulate, int Ntiles, int totalTiles)
{
    const int lane = threadIdx.x & 31;
    const int wave = threadIdx.x >> 5;
    const int tile = blockIdx.x * 8 + wave;
    if (tile >= totalTiles) return;           // wave-uniform; no barriers used

    const int tn   = tile % Ntiles;
    const int tm   = tile / Ntiles;
    const int lrow = lane & 15;
    const int hi   = lane >> 4;
    const int khA  = hi * 16;
    const int khB  = hi * 16;
    const int n    = tn * 16 + lrow;
    const int mb   = tm * 64;

    const f16* Ar[4];
#pragma unroll
    for (int s = 0; s < 4; ++s) {
        int mr = mb + s * 16 + lrow;
        int ar = mr;
        if (revL) { int b = mr / revL; int t = mr - b * revL; ar = b * revL + (revL - 1 - t); }
        Ar[s] = Ah + (size_t)ar * Kp;
    }
    const f16* Br = Bt + (size_t)n * Kp;

    v8f acc[4] = {};
    for (int k0 = 0; k0 < Kp; k0 += 32) {
        __builtin_prefetch(Br + k0 + 64, 0, 1);           // global_prefetch_b8
        v16h bf = *(const v16h*)(Br + k0 + khB);
#pragma unroll
        for (int s = 0; s < 4; ++s) {
            v8h a0 = *(const v8h*)(Ar[s] + k0 + khA);
            v8h a1 = *(const v8h*)(Ar[s] + k0 + 8 + khA);
            v16h af = __builtin_shufflevector(a0, a1,
                0, 1, 2, 3, 4, 5, 6, 7, 8, 9, 10, 11, 12, 13, 14, 15);
            acc[s] = __builtin_amdgcn_wmma_f32_16x16x32_f16(
                false, af, false, bf, (short)0, acc[s], false, false);
        }
    }

    const float bv = bias ? bias[n] : 0.0f;
#pragma unroll
    for (int s = 0; s < 4; ++s) {
#pragma unroll
        for (int r = 0; r < 8; ++r) {
            int m = mb + s * 16 + r + hi * 8;
            size_t idx = (size_t)m * N + n;
            float v = acc[s][r] + bv;
            if (accumulate) v += C[idx];
            C[idx] = v;
        }
    }
}

// ---------------------------------------------------------------------------
// LayerNorm over last dim D; writes f16 operand, optionally f32 too.
// ---------------------------------------------------------------------------
__global__ __launch_bounds__(256) void layernorm_k(
    const float* __restrict__ in, const float* __restrict__ w,
    const float* __restrict__ b, f16* __restrict__ out16,
    float* __restrict__ out32, int D)
{
    __shared__ float s1[256], s2[256];
    const float* x = in + (size_t)blockIdx.x * D;
    float sum = 0.f, sq = 0.f;
    for (int i = threadIdx.x; i < D; i += 256) { float v = x[i]; sum += v; sq += v * v; }
    s1[threadIdx.x] = sum; s2[threadIdx.x] = sq;
    __syncthreads();
    for (int s = 128; s > 0; s >>= 1) {
        if (threadIdx.x < s) { s1[threadIdx.x] += s1[threadIdx.x + s]; s2[threadIdx.x] += s2[threadIdx.x + s]; }
        __syncthreads();
    }
    float mean = s1[0] / D;
    float inv  = rsqrtf(s2[0] / D - mean * mean + kEps);
    for (int i = threadIdx.x; i < D; i += 256) {
        float v = (x[i] - mean) * inv * w[i] + b[i];
        out16[(size_t)blockIdx.x * D + i] = (f16)v;
        if (out32) out32[(size_t)blockIdx.x * D + i] = v;
    }
}

// ---------------------------------------------------------------------------
// Depthwise causal conv (width 7) + bias + SiLU, BOTH directions in one launch.
// Reads zxbcdt cols [1024,2080); writes xc{dir}[b,t,c].
// ---------------------------------------------------------------------------
__global__ __launch_bounds__(256) void conv_silu_k(
    const float* __restrict__ z0, const float* __restrict__ z1,
    const float* __restrict__ cw, const float* __restrict__ cb,
    float* __restrict__ xc0, float* __restrict__ xc1, int l)
{
    const int per = kBatch * kSeq * kConvDim;
    int idx = blockIdx.x * 256 + threadIdx.x;
    if (idx >= 2 * per) return;
    int dir = idx / per;
    int r   = idx - dir * per;
    int c = r % kConvDim;
    int t = (r / kConvDim) % kSeq;
    int b = r / (kConvDim * kSeq);

    const float* zx  = dir ? z1 : z0;
    float*       out = dir ? xc1 : xc0;
    const int    ld  = l * 2 + dir;
    const float* cwp = cw + (size_t)ld * kConvDim * kDConv;

    float acc = cb[(size_t)ld * kConvDim + c];
#pragma unroll
    for (int j = 0; j < kDConv; ++j) {
        int tt = t - (kDConv - 1) + j;
        if (tt >= 0)
            acc += zx[((size_t)(b * kSeq + tt)) * kDProj + kDInner + c] * cwp[c * kDConv + j];
    }
    out[r] = siluf(acc);
}

// ---------------------------------------------------------------------------
// Selective scan, BOTH directions: grid = 64 blocks (dir, batch, head),
// 64 lanes = head dim. Barrier-free: every lane loads B/C/dt itself
// (uniform addresses -> broadcast from L2); t+1 operands loaded before the
// t recurrence so latency overlaps the FMA chain. y written IN-PLACE into
// z cols [1024,2048) (dead after conv).
// ---------------------------------------------------------------------------
__global__ __launch_bounds__(64) void scan_k(
    const float* __restrict__ xc0, const float* __restrict__ xc1,
    float* __restrict__ z0, float* __restrict__ z1,
    const float* __restrict__ dtb, const float* __restrict__ Alog,
    const float* __restrict__ Dp, int l)
{
    const int dir = blockIdx.x >> 5;
    const int b   = (blockIdx.x >> 4) & 1;
    const int hd  = blockIdx.x & 15;
    const int p   = threadIdx.x;
    const int ld  = l * 2 + dir;

    const float* xbc = dir ? xc1 : xc0;
    float*       z   = dir ? z1 : z0;

    const float A   = -__expf(Alog[ld * kNHeads + hd]);
    const float Dv  = Dp[ld * kNHeads + hd];
    const float dtbias = dtb[ld * kNHeads + hd];

    const size_t row0 = (size_t)b * kSeq;
    const float* pB  = xbc + row0 * kConvDim + kDInner;              // +t*kConvDim
    const float* pX  = xbc + row0 * kConvDim + hd * kHeadDim + p;    // +t*kConvDim
    const float* pdt = z + row0 * kDProj + kDInner + kConvDim + hd;  // +t*kDProj
    float*       pY  = z + row0 * kDProj + kDInner + hd * kHeadDim + p; // +t*kDProj

    float st[kDState];
#pragma unroll
    for (int n = 0; n < kDState; ++n) st[n] = 0.f;

    // prime t = 0
    float4 nb0 = *(const float4*)(pB + 0),  nb1 = *(const float4*)(pB + 4);
    float4 nb2 = *(const float4*)(pB + 8),  nb3 = *(const float4*)(pB + 12);
    float4 nc0 = *(const float4*)(pB + 16), nc1 = *(const float4*)(pB + 20);
    float4 nc2 = *(const float4*)(pB + 24), nc3 = *(const float4*)(pB + 28);
    float  ndt = *pdt;
    float  nx  = *pX;

    for (int t = 0; t < kSeq; ++t) {
        float4 b0 = nb0, b1 = nb1, b2 = nb2, b3 = nb3;
        float4 c0 = nc0, c1 = nc1, c2 = nc2, c3 = nc3;
        float draw = ndt, xv = nx;

        if (t + 1 < kSeq) {               // issue next-step loads early
            const float* q = pB + (size_t)(t + 1) * kConvDim;
            nb0 = *(const float4*)(q + 0);  nb1 = *(const float4*)(q + 4);
            nb2 = *(const float4*)(q + 8);  nb3 = *(const float4*)(q + 12);
            nc0 = *(const float4*)(q + 16); nc1 = *(const float4*)(q + 20);
            nc2 = *(const float4*)(q + 24); nc3 = *(const float4*)(q + 28);
            ndt = pdt[(size_t)(t + 1) * kDProj];
            nx  = pX[(size_t)(t + 1) * kConvDim];
        }

        float d  = draw + dtbias;
        float dt = (d > 20.f) ? d : log1pf(__expf(d));   // softplus
        float dA = __expf(dt * A);
        float dtx = dt * xv;
        float acc = 0.f;
        st[0]  = st[0]  * dA + dtx * b0.x; acc += st[0]  * c0.x;
        st[1]  = st[1]  * dA + dtx * b0.y; acc += st[1]  * c0.y;
        st[2]  = st[2]  * dA + dtx * b0.z; acc += st[2]  * c0.z;
        st[3]  = st[3]  * dA + dtx * b0.w; acc += st[3]  * c0.w;
        st[4]  = st[4]  * dA + dtx * b1.x; acc += st[4]  * c1.x;
        st[5]  = st[5]  * dA + dtx * b1.y; acc += st[5]  * c1.y;
        st[6]  = st[6]  * dA + dtx * b1.z; acc += st[6]  * c1.z;
        st[7]  = st[7]  * dA + dtx * b1.w; acc += st[7]  * c1.w;
        st[8]  = st[8]  * dA + dtx * b2.x; acc += st[8]  * c2.x;
        st[9]  = st[9]  * dA + dtx * b2.y; acc += st[9]  * c2.y;
        st[10] = st[10] * dA + dtx * b2.z; acc += st[10] * c2.z;
        st[11] = st[11] * dA + dtx * b2.w; acc += st[11] * c2.w;
        st[12] = st[12] * dA + dtx * b3.x; acc += st[12] * c3.x;
        st[13] = st[13] * dA + dtx * b3.y; acc += st[13] * c3.y;
        st[14] = st[14] * dA + dtx * b3.z; acc += st[14] * c3.z;
        st[15] = st[15] * dA + dtx * b3.w; acc += st[15] * c3.w;

        pY[(size_t)t * kDProj] = acc + Dv * xv;
    }
}

// ---------------------------------------------------------------------------
// y = y * silu(z); RMS over 1024; scale by normw -> f16. Both dirs, one launch.
// y lives in z cols [1024,2048), gate z in cols [0,1024).
// ---------------------------------------------------------------------------
__global__ __launch_bounds__(256) void gatenorm_k(
    const float* __restrict__ z0, const float* __restrict__ z1,
    const float* __restrict__ normw, f16* __restrict__ yn0,
    f16* __restrict__ yn1, int l)
{
    __shared__ float sr[256];
    const int dir = (blockIdx.x >= kTokens) ? 1 : 0;
    const size_t row = blockIdx.x - dir * kTokens;
    const float* zx = (dir ? z1 : z0) + row * kDProj;
    f16* yn = (dir ? yn1 : yn0) + row * kDInner;
    const float* nw = normw + (size_t)(l * 2 + dir) * kDInner;

    float vals[4];
    float ss = 0.f;
#pragma unroll
    for (int i = 0; i < 4; ++i) {
        int c = threadIdx.x + i * 256;
        float zv = zx[c];
        float v  = zx[kDInner + c] * siluf(zv);
        vals[i] = v;
        ss += v * v;
    }
    sr[threadIdx.x] = ss;
    __syncthreads();
    for (int s = 128; s > 0; s >>= 1) {
        if (threadIdx.x < s) sr[threadIdx.x] += sr[threadIdx.x + s];
        __syncthreads();
    }
    float inv = rsqrtf(sr[0] / (float)kDInner + kEps);
#pragma unroll
    for (int i = 0; i < 4; ++i) {
        int c = threadIdx.x + i * 256;
        yn[c] = (f16)(vals[i] * inv * nw[c]);
    }
}

// ---------------------------------------------------------------------------
extern "C" void kernel_launch(void* const* d_in, const int* in_sizes, int n_in,
                              void* d_out, int out_size, void* d_ws, size_t ws_size,
                              hipStream_t stream)
{
    (void)in_sizes; (void)n_in; (void)out_size; (void)ws_size;
    const float* x     = (const float*)d_in[0];
    const float* Wpi   = (const float*)d_in[1];
    const float* bpi   = (const float*)d_in[2];
    const float* ln_w  = (const float*)d_in[3];
    const float* ln_b  = (const float*)d_in[4];
    const float* Win   = (const float*)d_in[5];
    const float* convw = (const float*)d_in[6];
    const float* convb = (const float*)d_in[7];
    const float* dtb   = (const float*)d_in[8];
    const float* Alog  = (const float*)d_in[9];
    const float* Dp    = (const float*)d_in[10];
    const float* normw = (const float*)d_in[11];
    const float* Wout  = (const float*)d_in[12];
    const float* fn_w  = (const float*)d_in[13];
    const float* fn_b  = (const float*)d_in[14];
    const float* Wpo   = (const float*)d_in[15];
    const float* bpo   = (const float*)d_in[16];
    float* out = (float*)d_out;

    // ---- workspace bump allocator (256B aligned regions) ----
    char* base = (char*)d_ws;
    size_t cur = 0;
    auto alloc = [&](size_t bytes) -> void* {
        void* p = base + cur;
        cur += (bytes + 255) & ~(size_t)255;
        return p;
    };
    float* hA   = (float*)alloc((size_t)kTokens * kDModel * 4);
    float* hB   = (float*)alloc((size_t)kTokens * kDModel * 4);
    float* z0   = (float*)alloc((size_t)kTokens * kDProj * 4);
    float* z1   = (float*)alloc((size_t)kTokens * kDProj * 4);
    float* xc0  = (float*)alloc((size_t)kTokens * kConvDim * 4);
    float* xc1  = (float*)alloc((size_t)kTokens * kConvDim * 4);
    f16*   hn16 = (f16*)alloc((size_t)kTokens * kDModel * 2);
    f16*   yn0  = (f16*)alloc((size_t)kTokens * kDInner * 2);
    f16*   yn1  = (f16*)alloc((size_t)kTokens * kDInner * 2);
    f16*   xh   = (f16*)alloc((size_t)kTokens * 96 * 2);       // K=80 -> Kp=96
    f16*   WpiT = (f16*)alloc((size_t)kDModel * 96 * 2);
    f16*   WinT = (f16*)alloc((size_t)4 * kDProj * kDModel * 2);
    f16*   WoutT= (f16*)alloc((size_t)4 * kDModel * kDInner * 2);
    f16*   WpoT = (f16*)alloc((size_t)kFeat * kDModel * 2);

    const dim3 blk(256);
    auto cdiv = [](int a, int b) { return (a + b - 1) / b; };

    // ---- pack weights (transposed, K-padded f16) & input ----
    pack_transpose_f16<<<cdiv(kDModel * 96, 256), blk, 0, stream>>>(
        Wpi, WpiT, kFeat, kDModel, 96, kDModel * 96);
    for (int ld = 0; ld < 4; ++ld) {
        pack_transpose_f16<<<cdiv(kDProj * kDModel, 256), blk, 0, stream>>>(
            Win + (size_t)ld * kDModel * kDProj, WinT + (size_t)ld * kDProj * kDModel,
            kDModel, kDProj, kDModel, kDProj * kDModel);
        pack_transpose_f16<<<cdiv(kDModel * kDInner, 256), blk, 0, stream>>>(
            Wout + (size_t)ld * kDInner * kDModel, WoutT + (size_t)ld * kDModel * kDInner,
            kDInner, kDModel, kDInner, kDModel * kDInner);
    }
    pack_transpose_f16<<<cdiv(kFeat * kDModel, 256), blk, 0, stream>>>(
        Wpo, WpoT, kDModel, kFeat, kDModel, kFeat * kDModel);
    pack_rowmajor_f16<<<cdiv(kTokens * 96, 256), blk, 0, stream>>>(
        x, xh, kFeat, 96, kTokens * 96);

    auto gemm = [&](const f16* A, const f16* B, const float* bias, float* C,
                    int N, int Kp, int rev, int accum) {
        int Ntiles = N / 16;
        int total  = (kTokens / 64) * Ntiles;
        wmma_gemm<<<cdiv(total, 8), blk, 0, stream>>>(
            A, B, bias, C, N, Kp, rev, accum, Ntiles, total);
    };

    // h = x @ Wpi + bpi
    gemm(xh, WpiT, bpi, hA, kDModel, 96, 0, 0);

    float* hCur = hA;
    float* hNext = hB;
    const size_t hBytes = (size_t)kTokens * kDModel * sizeof(float);

    for (int l = 0; l < 2; ++l) {
        layernorm_k<<<kTokens, blk, 0, stream>>>(
            hCur, ln_w + l * kDModel, ln_b + l * kDModel, hn16, nullptr, kDModel);
        hipMemcpyAsync(hNext, hCur, hBytes, hipMemcpyDeviceToDevice, stream); // residual

        // in-projections, both directions (dir1 = time-reversed A rows)
        gemm(hn16, WinT + (size_t)(l * 2 + 0) * kDProj * kDModel, nullptr, z0,
             kDProj, kDModel, 0, 0);
        gemm(hn16, WinT + (size_t)(l * 2 + 1) * kDProj * kDModel, nullptr, z1,
             kDProj, kDModel, kSeq, 0);

        conv_silu_k<<<cdiv(2 * kBatch * kSeq * kConvDim, 256), blk, 0, stream>>>(
            z0, z1, convw, convb, xc0, xc1, l);

        scan_k<<<2 * kBatch * kNHeads, dim3(64), 0, stream>>>(
            xc0, xc1, z0, z1, dtb, Alog, Dp, l);

        gatenorm_k<<<2 * kTokens, blk, 0, stream>>>(
            z0, z1, normw, yn0, yn1, l);

        // out-projections accumulate into residual
        gemm(yn0, WoutT + (size_t)(l * 2 + 0) * kDModel * kDInner, nullptr, hNext,
             kDModel, kDInner, 0, 1);
        gemm(yn1, WoutT + (size_t)(l * 2 + 1) * kDModel * kDInner, nullptr, hNext,
             kDModel, kDInner, kSeq, 1);

        float* tmp = hCur; hCur = hNext; hNext = tmp;
    }

    // final layernorm: f16 operand for pred GEMM + f32 h-output written in place
    layernorm_k<<<kTokens, blk, 0, stream>>>(
        hCur, fn_w, fn_b, hn16, out + (size_t)kTokens * kFeat, kDModel);

    // pred = hn @ Wpo + bpo
    gemm(hn16, WpoT, bpo, out, kFeat, kDModel, 0, 0);
}
